// Attention_45784351375747
// MI455X (gfx1250) — matrix operand order repaired and checked by reading, not compiled, not run on hardware
//
#include <hip/hip_runtime.h>
#include <hip/hip_bf16.h>

// ---------------------------------------------------------------------------
// Causal single-head attention for MI455X (gfx1250), bf16 WMMA + fp32 accum.
//   Kernel 1: q/k/v = x @ W{q,k,v}  (f32 in, bf16 out; V stored TRANSPOSED
//             [H][M] so the attention kernel never has to transpose it)
//   Kernel 2: flash attention, O register-resident via H-split across waves,
//             K/V tiles DMA'd into LDS by the Tensor Data Mover (TENSORcnt),
//             S tile never touches HBM.
// ---------------------------------------------------------------------------

typedef __attribute__((ext_vector_type(16))) __bf16 v16bf;
typedef __attribute__((ext_vector_type(8)))  float  v8f;
typedef __attribute__((ext_vector_type(4)))  unsigned int v4u;
typedef __attribute__((ext_vector_type(8)))  int    v8i;
typedef __attribute__((ext_vector_type(4)))  int    v4i;

#define B_ 8
#define T_ 2048
#define D_ 1024
#define H_ 1024
#define M_ (B_ * T_)  // 16384

#if __has_builtin(__builtin_amdgcn_tensor_load_to_lds) && \
    __has_builtin(__builtin_amdgcn_s_wait_tensorcnt)
#define USE_TDM 1
#else
#define USE_TDM 0
#endif

__device__ __forceinline__ unsigned short f2bf(float f) {
  unsigned int u = __float_as_uint(f);
  u += 0x7FFFu + ((u >> 16) & 1u);      // round-to-nearest-even
  return (unsigned short)(u >> 16);
}

union FragU { uint4 q[2]; v16bf v; };
union AccU  { v8f  v; float f[8]; };

// A fragment: 16xK row-major bf16 tile (stride in ushort elements).
// Lane L row M=L&15; 16 bf16 at K offsets [8*kh,8*kh+8) and [16+8*kh,+8).
__device__ __forceinline__ v16bf load_a_frag(const unsigned short* p, int stride, int lane) {
  const int row = lane & 15, kh = lane >> 4;
  const unsigned short* base = p + row * stride + 8 * kh;
  FragU a;
  a.q[0] = *(const uint4*)(base);
  a.q[1] = *(const uint4*)(base + 16);
  return a.v;
}

// B fragment: KxN matrix stored N-major in LDS (p[n*stride + k]).
// Lane L column N=L&15; 16 bf16 at K offsets [8*kh,8*kh+8) and [16+8*kh,+8).
__device__ __forceinline__ v16bf load_b_frag(const unsigned short* p, int stride, int lane) {
  const int col = lane & 15, kh = lane >> 4;
  const unsigned short* base = p + col * stride + 8 * kh;
  FragU b;
  b.q[0] = *(const uint4*)(base);
  b.q[1] = *(const uint4*)(base + 16);
  return b.v;
}

#if USE_TDM
// Tensor Data Mover: 2D bf16 tile (tile_d0 elems/row x tile_d1 rows) from
// global (row stride = stride0 elems) into LDS, packed linearly.
// D# built per CDNA5 ISA 8.3/8.4: group0 = {flags, lds_addr, global_addr, type=2},
// group1 = {data_size=2B, tensor dims, tile dims, dim0 stride}. 2D -> groups 2/3 zero.
// This toolchain exposes the 6-arg builtin: (v4u, v8i, v4i, v4i, v8i, imm cpol).
__device__ __forceinline__ void tdm_load_2d_bf16(const unsigned short* gsrc,
                                                 unsigned lds_byte_off,
                                                 unsigned tile_d0, unsigned tile_d1,
                                                 unsigned stride0) {
  unsigned long long ga = (unsigned long long)(uintptr_t)gsrc;
  v4u g0;
  g0.x = 1u;                                              // count=1 (valid), user mode
  g0.y = lds_byte_off;                                    // lds_addr [63:32]
  g0.z = (unsigned)(ga & 0xFFFFFFFFu);                    // global_addr [95:64]
  g0.w = (unsigned)((ga >> 32) & 0x01FFFFFFu) | (2u << 30); // addr[56:32] | type=2
  v8i g1;
  g1[0] = (int)(1u << 16);                                // data_size=1 -> 2 bytes
  g1[1] = (int)((tile_d0 & 0xFFFFu) << 16);               // tensor_dim0[15:0]
  g1[2] = (int)(((tile_d0 >> 16) & 0xFFFFu) | ((tile_d1 & 0xFFFFu) << 16)); // dim0 hi | dim1 lo
  g1[3] = (int)(((tile_d1 >> 16) & 0xFFFFu) | ((tile_d0 & 0xFFFFu) << 16)); // dim1 hi | tile_dim0
  g1[4] = (int)(tile_d1 & 0xFFFFu);                       // tile_dim1 (tile_dim2=0)
  g1[5] = (int)stride0;                                   // tensor_dim0_stride[31:0]
  g1[6] = 0;                                              // stride0[47:32], dim1_stride lo
  g1[7] = 0;                                              // dim1_stride hi
  v4i z4 = {0, 0, 0, 0};
  v8i z8 = {0, 0, 0, 0, 0, 0, 0, 0};
  __builtin_amdgcn_tensor_load_to_lds(g0, g1, z4, z4, z8, 0);
}
__device__ __forceinline__ unsigned lds_offset_of(const void* p) {
  return (unsigned)(uintptr_t)p;   // low 32 bits of a generic LDS pointer = LDS offset
}
#endif

// ---------------------------------------------------------------------------
// Kernel 1: QKV projection GEMM.  grid = (M/128, H/128, 3), block = 256 (8 waves)
// Block tile 128x128, wave tile 32x64 (2x4 wmma tiles), K step 32.
// z==2 (V) is stored transposed: vbT[h * M + m].
// ---------------------------------------------------------------------------
__global__ __launch_bounds__(256)
void qkv_gemm_kernel(const float* __restrict__ x,
                     const float* __restrict__ Wq,
                     const float* __restrict__ Wk,
                     const float* __restrict__ Wv,
                     unsigned short* __restrict__ qb,
                     unsigned short* __restrict__ kb,
                     unsigned short* __restrict__ vbT) {
  __shared__ unsigned short xs[128 * 32];   // [m][k] bf16
  __shared__ unsigned short wt[128 * 32];   // [n][k] bf16 (transposed on store)

  const int m0 = blockIdx.x * 128;
  const int n0 = blockIdx.y * 128;
  const float* W = (blockIdx.z == 0) ? Wq : (blockIdx.z == 1) ? Wk : Wv;
  unsigned short* out = (blockIdx.z == 0) ? qb : (blockIdx.z == 1) ? kb : vbT;
  const bool transposed_out = (blockIdx.z == 2);

  const int tid  = threadIdx.x;
  const int lane = tid & 31;
  const int wave = tid >> 5;
  const int wm   = wave >> 1;   // 0..3 -> 32-row group
  const int wn   = wave & 1;    // 0..1 -> 64-col group

  v8f acc[2][4] = {};

  for (int k0 = 0; k0 < D_; k0 += 32) {
    // Stage x tile (128x32) with f32->bf16 conversion.
#pragma unroll
    for (int i = 0; i < 16; ++i) {
      int l = tid + i * 256;                 // 0..4095
      int r = l >> 5, c = l & 31;
      xs[r * 32 + c] = f2bf(x[(size_t)(m0 + r) * D_ + (k0 + c)]);
    }
    // Stage W tile (32x128), transposed to [n][k].
#pragma unroll
    for (int i = 0; i < 16; ++i) {
      int l = tid + i * 256;
      int k = l >> 7, n = l & 127;
      wt[n * 32 + k] = f2bf(W[(size_t)(k0 + k) * H_ + (n0 + n)]);
    }
    if (k0 + 32 < D_) {                      // gfx1250 global_prefetch_b8
      __builtin_prefetch((const void*)&x[(size_t)(m0 + (tid & 127)) * D_ + k0 + 32], 0, 1);
      __builtin_prefetch((const void*)&W[(size_t)(k0 + 32 + (tid & 31)) * H_ + n0], 0, 1);
    }
    __syncthreads();

    v16bf afr[2], bfr[4];
#pragma unroll
    for (int mt = 0; mt < 2; ++mt)
      afr[mt] = load_a_frag(&xs[(wm * 32 + mt * 16) * 32], 32, lane);
#pragma unroll
    for (int nt = 0; nt < 4; ++nt)
      bfr[nt] = load_b_frag(&wt[(wn * 64 + nt * 16) * 32], 32, lane);
#pragma unroll
    for (int mt = 0; mt < 2; ++mt)
#pragma unroll
      for (int nt = 0; nt < 4; ++nt)
        acc[mt][nt] = __builtin_amdgcn_wmma_f32_16x16x32_bf16(
            false, afr[mt], false, bfr[nt], (short)0, acc[mt][nt], false, false);
    __syncthreads();
  }

  // Epilogue: C layout M = v + 8*(lane>>4), N = lane&15.  Store bf16.
#pragma unroll
  for (int mt = 0; mt < 2; ++mt)
#pragma unroll
    for (int nt = 0; nt < 4; ++nt) {
      AccU a; a.v = acc[mt][nt];
      const int gn = n0 + wn * 64 + nt * 16 + (lane & 15);
      const int mbase = m0 + wm * 32 + mt * 16 + ((lane >> 4) << 3);
#pragma unroll
      for (int vv = 0; vv < 8; ++vv) {
        unsigned short bfv = f2bf(a.f[vv]);
        if (transposed_out) out[(size_t)gn * M_ + (mbase + vv)] = bfv;   // v^T [H][M]
        else                out[(size_t)(mbase + vv) * H_ + gn] = bfv;   // q/k [M][H]
      }
    }
}

// ---------------------------------------------------------------------------
// Kernel 2: flash attention.  grid = (T/16, B), block = 256 (8 waves).
// 16 query rows per block; each wave owns a 128-wide H slice of O (8 C frags,
// register resident).  K tile [key][h] and V^T tile [h][key] staged in LDS via
// the Tensor Data Mover (fallback: uint4 copies).  ~163KB dynamic LDS.
// ---------------------------------------------------------------------------
__global__ __launch_bounds__(256)
void flash_attn_kernel(const unsigned short* __restrict__ qb,
                       const unsigned short* __restrict__ kb,
                       const unsigned short* __restrict__ vbT,
                       float* __restrict__ out) {
  extern __shared__ __align__(16) unsigned char smem_raw[];
  unsigned short* Qs  = (unsigned short*)smem_raw;   // 16  x 1024 bf16 (32KB)
  unsigned short* Ks  = Qs + 16 * 1024;              // 32  x 1024 bf16 (64KB) [key][h]
  unsigned short* Vst = Ks + 32 * 1024;              // 1024 x 32 bf16 (64KB)  [h][key]
  float*          Sred   = (float*)(Vst + 1024 * 32);     // 16 x 32 f32
  unsigned short* Pt     = (unsigned short*)(Sred + 512); // 16 x 32 bf16
  float*          alphas = (float*)(Pt + 512);            // 16
  float*          rowm   = alphas + 16;                   // 16
  float*          rowl   = rowm + 16;                     // 16

  const int b    = blockIdx.y;
  const int q0   = blockIdx.x * 16;
  const int tid  = threadIdx.x;
  const int lane = tid & 31;
  const int wave = tid >> 5;
  const int h_base = wave * 128;

  // Load the Q tile once (16 x 1024 bf16).
#if USE_TDM
  if (wave == 0) {
    tdm_load_2d_bf16(qb + (size_t)(b * T_ + q0) * H_, lds_offset_of(Qs),
                     /*tile_d0=*/H_, /*tile_d1=*/16, /*stride0=*/H_);
    __builtin_amdgcn_s_wait_tensorcnt(0);
  }
#else
  {
    const uint4* src = (const uint4*)(qb + (size_t)(b * T_ + q0) * H_);
    uint4* dst = (uint4*)Qs;
#pragma unroll
    for (int i = 0; i < 8; ++i) dst[tid + i * 256] = src[tid + i * 256];
  }
#endif
  if (tid < 16) { rowm[tid] = -3.0e38f; rowl[tid] = 0.0f; }

  v8f ofr[8] = {};
  __syncthreads();

  for (int k0 = 0; k0 < q0 + 16; k0 += 32) {   // causal: tiles at/below diagonal
#if USE_TDM
    // TDM DMA: wave0 pulls K tile (32 keys x 1024), wave1 pulls V^T tile
    // (1024 h-rows x 32 keys).  TENSORcnt tracks completion; barrier publishes.
    if (wave == 0) {
      tdm_load_2d_bf16(kb + (size_t)(b * T_ + k0) * H_, lds_offset_of(Ks),
                       /*tile_d0=*/H_, /*tile_d1=*/32, /*stride0=*/H_);
      __builtin_amdgcn_s_wait_tensorcnt(0);
    } else if (wave == 1) {
      tdm_load_2d_bf16(vbT + (size_t)(b * T_) + k0, lds_offset_of(Vst),
                       /*tile_d0=*/32, /*tile_d1=*/H_, /*stride0=*/M_);
      __builtin_amdgcn_s_wait_tensorcnt(0);
    }
#else
    {  // K tile [key][h]
      const uint4* src = (const uint4*)(kb + (size_t)(b * T_ + k0) * H_);
      uint4* dst = (uint4*)Ks;
#pragma unroll
      for (int i = 0; i < 16; ++i) dst[tid + i * 256] = src[tid + i * 256];
    }
    {  // V^T tile [h][key]: rows of v^T are key-contiguous -> pure uint4 copies
#pragma unroll
      for (int i = 0; i < 16; ++i) {
        int l = tid + i * 256;                 // 0..4095 uint4s
        int h = l >> 2, part = l & 3;
        const uint4* src = (const uint4*)(vbT + (size_t)h * M_ + b * T_ + k0);
        ((uint4*)Vst)[l] = src[part];
      }
    }
#endif
    Sred[tid] = 0.0f; Sred[tid + 256] = 0.0f;
    __syncthreads();

    // Partial S = Q . K^T over this wave's 128-wide H slice (4 K-steps of 32).
    AccU sacc[2]; sacc[0].v = (v8f){}; sacc[1].v = (v8f){};
#pragma unroll
    for (int ks = 0; ks < 4; ++ks) {
      const int h0 = h_base + ks * 32;
      v16bf aq = load_a_frag(Qs + h0, 1024, lane);
#pragma unroll
      for (int nt = 0; nt < 2; ++nt) {
        v16bf bk = load_b_frag(Ks + (size_t)(nt * 16) * 1024 + h0, 1024, lane);
        sacc[nt].v = __builtin_amdgcn_wmma_f32_16x16x32_bf16(
            false, aq, false, bk, (short)0, sacc[nt].v, false, false);
      }
    }
    // Cross-wave reduction of S into LDS via ds_add_f32.
#pragma unroll
    for (int nt = 0; nt < 2; ++nt)
#pragma unroll
      for (int vv = 0; vv < 8; ++vv) {
        int Mr = vv + ((lane >> 4) << 3);
        int Nc = nt * 16 + (lane & 15);
        atomicAdd(&Sred[Mr * 32 + Nc], sacc[nt].f[vv]);
      }
    __syncthreads();

    // Online softmax: wave 0, lanes 0..15 each own one query row.
    if (wave == 0 && lane < 16) {
      const int r = lane;
      float mold = rowm[r];
      float mmax = mold;
      float sv[32];
#pragma unroll
      for (int j = 0; j < 32; ++j) {
        float s = Sred[r * 32 + j] * 0.03125f;         // 1/sqrt(H), H=1024
        if (k0 + j > q0 + r) s = -3.0e38f;             // causal mask
        sv[j] = s;
        mmax = fmaxf(mmax, s);
      }
      float alpha = __expf(mold - mmax);
      float lsum  = rowl[r] * alpha;
#pragma unroll
      for (int j = 0; j < 32; ++j) {
        float p = __expf(sv[j] - mmax);                // masked -> exp(-huge)=0
        lsum += p;
        Pt[r * 32 + j] = f2bf(p);
      }
      rowm[r] = mmax; rowl[r] = lsum; alphas[r] = alpha;
    }
    __syncthreads();

    // Rescale O by alpha[row], then O += P . V over this wave's H slice.
    {
      const int kh = lane >> 4;
      float ar[8];
#pragma unroll
      for (int vv = 0; vv < 8; ++vv) ar[vv] = alphas[vv + (kh << 3)];
#pragma unroll
      for (int nt = 0; nt < 8; ++nt) {
        AccU o; o.v = ofr[nt];
#pragma unroll
        for (int vv = 0; vv < 8; ++vv) o.f[vv] *= ar[vv];
        ofr[nt] = o.v;
      }
      v16bf ap = load_a_frag(Pt, 32, lane);
#pragma unroll
      for (int nt = 0; nt < 8; ++nt) {
        v16bf bv = load_b_frag(Vst + (size_t)(h_base + nt * 16) * 32, 32, lane);
        ofr[nt] = __builtin_amdgcn_wmma_f32_16x16x32_bf16(
            false, ap, false, bv, (short)0, ofr[nt], false, false);
      }
    }
    __syncthreads();   // before next iteration overwrites Ks/Vst
  }

  // Normalize by the softmax denominator and store fp32 output.
  {
    const int kh = lane >> 4;
    float linv[8];
#pragma unroll
    for (int vv = 0; vv < 8; ++vv) {
      float l = rowl[vv + (kh << 3)];
      linv[vv] = (l > 0.0f) ? (1.0f / l) : 0.0f;
    }
#pragma unroll
    for (int nt = 0; nt < 8; ++nt) {
      AccU o; o.v = ofr[nt];
      const int gh = h_base + nt * 16 + (lane & 15);
#pragma unroll
      for (int vv = 0; vv < 8; ++vv) {
        const int gq = q0 + vv + (kh << 3);
        out[(size_t)(b * T_ + gq) * H_ + gh] = o.f[vv] * linv[vv];
      }
    }
  }
}

// ---------------------------------------------------------------------------
extern "C" void kernel_launch(void* const* d_in, const int* in_sizes, int n_in,
                              void* d_out, int out_size, void* d_ws, size_t ws_size,
                              hipStream_t stream) {
  const float* x  = (const float*)d_in[0];
  const float* Wq = (const float*)d_in[1];
  const float* Wk = (const float*)d_in[2];
  const float* Wv = (const float*)d_in[3];
  float* out = (float*)d_out;

  // Workspace: q,k as [M][H] bf16, v as [H][M] bf16; 3 * 33.5MB = 100.7MB.
  unsigned short* qb  = (unsigned short*)d_ws;
  unsigned short* kb  = qb + (size_t)M_ * H_;
  unsigned short* vbT = kb + (size_t)M_ * H_;

  qkv_gemm_kernel<<<dim3(M_ / 128, H_ / 128, 3), 256, 0, stream>>>(
      x, Wq, Wk, Wv, qb, kb, vbT);

  const size_t lds_bytes =
      (16 * 1024 + 32 * 1024 + 1024 * 32) * sizeof(unsigned short)  // Qs,Ks,Vst
      + 16 * 32 * sizeof(float)                                     // Sred
      + 16 * 32 * sizeof(unsigned short)                            // Pt
      + 48 * sizeof(float);                                         // alphas/rowm/rowl
  // ~163KB dynamic LDS: within CDNA5's 320KB-per-workgroup limit.
  flash_attn_kernel<<<dim3(T_ / 16, B_, 1), 256, lds_bytes, stream>>>(
      qb, kb, vbT, out);
}
